// SymbolRetriever_3015067041951
// MI455X (gfx1250) — matrix operand order repaired and checked by reading, not compiled, not run on hardware
//
#include <hip/hip_runtime.h>
#include <hip/hip_bf16.h>
#include <math.h>

// ---------------------------------------------------------------------------
// SymbolRetriever fused kernel for gfx1250 (MI455X), wave32 + WMMA bf16.
//   out = softmax( (X @ Wq) @ binding^T * (1/32) ) @ symbols
// X:[32768,1024] f32, Wq:[1024,1024], binding:[512,1024], symbols:[512,1024]
//
// Per workgroup (8 waves / 256 threads): 64 query rows, fully fused in LDS.
//   Stage X : X tile f32 -> bf16, staged once into LDS          (coalesced)
//   Stage A : Q  = Xs @ Wq          -> LDS (bf16)               (WMMA)
//   Stage B : S  = scale * Q @ B^T  -> LDS (f32)                (WMMA)
//   softmax : S -> attn (bf16, overwrites dead Q region)
//   Stage C : Out = attn @ symbols  -> HBM                      (WMMA)
// Each wave computes 64x64 macro-tiles: 16 WMMAs per k-step fed by 4 LDS-A
// fragments and 4 global-B fragments that are double-buffered one k-step
// ahead (B latency hidden behind a full 16-WMMA chain).
// ---------------------------------------------------------------------------

typedef __bf16 bf16_t;
typedef __attribute__((ext_vector_type(16))) __bf16 v16bf;
typedef __attribute__((ext_vector_type(8)))  __bf16 v8bf;
typedef __attribute__((ext_vector_type(4)))  __bf16 v4bf;
typedef __attribute__((ext_vector_type(8)))  float  v8f;
typedef __attribute__((ext_vector_type(4)))  float  v4f;

#define D_DIM   1024      // model dim
#define J_DIM   512       // n_symbols
#define M_TILE  64        // query rows per workgroup
#define NWAVES  8         // 256 threads
#define SM_SCALE 0.03125f // 1/sqrt(1024)

// Padded LDS strides (elements) to spread rows across the 64 banks.
#define QS_LD   (D_DIM + 8)   // bf16: row stride 2064B -> 4-bank shift/row
#define SS_LD   (J_DIM + 8)   // f32 : row stride 2080B -> 8-bank shift/row
#define AS_LD   (J_DIM + 8)   // bf16: row stride 1040B -> 4-bank shift/row

#define R0_BYTES (M_TILE * QS_LD * 2)   // 132096: Qs, later attn
#define R1_BYTES (M_TILE * SS_LD * 4)   // 133120: Xs (bf16), later Ss (f32)
#define LDS_BYTES (R0_BYTES + R1_BYTES) // 265216 < 320KB WGP budget

__device__ __forceinline__ v8f wmma_bf16(v16bf a, v16bf b, v8f c) {
  return __builtin_amdgcn_wmma_f32_16x16x32_bf16(
      false, a, false, b, (short)0, c, false, false);
}

// A-fragment (16x32 bf16) from bf16 row-major LDS.
// A layout: lane<16 -> row=lane,    K = {k0..+15}
//           lane>=16 -> row=lane-16, K = {k0+16..+31}
__device__ __forceinline__ v16bf load_a_lds(const bf16_t* base,
                                            int ld, int row0, int k0, int lane) {
  const int r   = row0 + (lane & 15);
  const int off = k0 + ((lane & 16) ? 16 : 0);
  const bf16_t* p = base + r * ld + off;
  v8bf lo = *(const v8bf*)(p);
  v8bf hi = *(const v8bf*)(p + 8);
  v16bf f;
#pragma unroll
  for (int i = 0; i < 8; ++i) { f[i] = lo[i]; f[8 + i] = hi[i]; }
  return f;
}

// B-fragment (32x16 bf16) from *N-major* bf16 global memory (mem[n][k]).
__device__ __forceinline__ v16bf load_b_nmajor(const bf16_t* __restrict__ base,
                                               int ld, int k0, int n0, int lane) {
  const bf16_t* p = base + (size_t)(n0 + (lane & 15)) * ld
                         + k0 + ((lane & 16) ? 16 : 0);
  return *(const v16bf*)p;
}

// 64x64 macro-tile GEMM: A [64][KDIM] in LDS, B N-major in global.
// acc[mi + 4*ni] covers rows mi*16.., cols n0 + ni*16..
// B fragments are double-buffered one k-step ahead.
template <int KDIM>
__device__ __forceinline__ void gemm_macrotile(const bf16_t* Abase, int Ald,
                                               const bf16_t* __restrict__ Bbase,
                                               int Bld, int n0, int lane,
                                               v8f acc[16]) {
  v16bf b0 = load_b_nmajor(Bbase, Bld, 0, n0,      lane);
  v16bf b1 = load_b_nmajor(Bbase, Bld, 0, n0 + 16, lane);
  v16bf b2 = load_b_nmajor(Bbase, Bld, 0, n0 + 32, lane);
  v16bf b3 = load_b_nmajor(Bbase, Bld, 0, n0 + 48, lane);
  for (int k0 = 0; k0 < KDIM; k0 += 32) {
    v16bf nb0 = b0, nb1 = b1, nb2 = b2, nb3 = b3;
    if (k0 + 32 < KDIM) {  // issue next k-step's B loads up front
      nb0 = load_b_nmajor(Bbase, Bld, k0 + 32, n0,      lane);
      nb1 = load_b_nmajor(Bbase, Bld, k0 + 32, n0 + 16, lane);
      nb2 = load_b_nmajor(Bbase, Bld, k0 + 32, n0 + 32, lane);
      nb3 = load_b_nmajor(Bbase, Bld, k0 + 32, n0 + 48, lane);
      if (k0 + 64 < KDIM) {  // deep prefetch -> global_prefetch_b8
#pragma unroll
        for (int ni = 0; ni < 4; ++ni)
          __builtin_prefetch(
              Bbase + (size_t)(n0 + ni * 16 + (lane & 15)) * Bld + k0 + 64, 0, 1);
      }
    }
    v16bf a0 = load_a_lds(Abase, Ald, 0,  k0, lane);
    v16bf a1 = load_a_lds(Abase, Ald, 16, k0, lane);
    v16bf a2 = load_a_lds(Abase, Ald, 32, k0, lane);
    v16bf a3 = load_a_lds(Abase, Ald, 48, k0, lane);
    acc[0]  = wmma_bf16(a0, b0, acc[0]);
    acc[1]  = wmma_bf16(a1, b0, acc[1]);
    acc[2]  = wmma_bf16(a2, b0, acc[2]);
    acc[3]  = wmma_bf16(a3, b0, acc[3]);
    acc[4]  = wmma_bf16(a0, b1, acc[4]);
    acc[5]  = wmma_bf16(a1, b1, acc[5]);
    acc[6]  = wmma_bf16(a2, b1, acc[6]);
    acc[7]  = wmma_bf16(a3, b1, acc[7]);
    acc[8]  = wmma_bf16(a0, b2, acc[8]);
    acc[9]  = wmma_bf16(a1, b2, acc[9]);
    acc[10] = wmma_bf16(a2, b2, acc[10]);
    acc[11] = wmma_bf16(a3, b2, acc[11]);
    acc[12] = wmma_bf16(a0, b3, acc[12]);
    acc[13] = wmma_bf16(a1, b3, acc[13]);
    acc[14] = wmma_bf16(a2, b3, acc[14]);
    acc[15] = wmma_bf16(a3, b3, acc[15]);
    b0 = nb0; b1 = nb1; b2 = nb2; b3 = nb3;
  }
}

// ---------------------------------------------------------------------------
// Kernel 0: build bf16 N-major weight copies in workspace.
//   WqT  [1024 n][1024 k] <- Wq[k][n]
//   BindB[512 j][1024 k]  <- binding[j][k]   (already N-major)
//   SymT [1024 n][512 j]  <- symbols[j][n]
// ---------------------------------------------------------------------------
__global__ void symret_convert(const float* __restrict__ Wq,
                               const float* __restrict__ binding,
                               const float* __restrict__ symbols,
                               bf16_t* __restrict__ WqT,
                               bf16_t* __restrict__ BindB,
                               bf16_t* __restrict__ SymT) {
  const int i = blockIdx.x * blockDim.x + threadIdx.x;
  if (i < D_DIM * D_DIM) {
    const int n = i >> 10, k = i & 1023;
    WqT[i] = (bf16_t)Wq[(size_t)k * D_DIM + n];
  }
  if (i < J_DIM * D_DIM) {
    BindB[i] = (bf16_t)binding[i];
  }
  if (i < D_DIM * J_DIM) {
    const int n = i / J_DIM, j = i - n * J_DIM;
    SymT[i] = (bf16_t)symbols[(size_t)j * D_DIM + n];
  }
}

// ---------------------------------------------------------------------------
// Fused kernel: one workgroup = 64 query rows, 8 waves, 259KB dynamic LDS.
// ---------------------------------------------------------------------------
__global__ void __launch_bounds__(256, 1)
symret_fused(const float* __restrict__ X,
             const bf16_t* __restrict__ WqT,
             const bf16_t* __restrict__ BindB,
             const bf16_t* __restrict__ SymT,
             float* __restrict__ Out) {
  extern __shared__ char smem[];
  bf16_t* Qs = (bf16_t*)smem;                 // [64][QS_LD] bf16 (region 0)
  bf16_t* As = (bf16_t*)smem;                 // attn bf16, aliases dead Qs
  bf16_t* Xs = (bf16_t*)(smem + R0_BYTES);    // [64][QS_LD] bf16 (region 1)
  float*  Ss = (float*)(smem + R0_BYTES);     // [64][SS_LD] f32, aliases Xs

  const int tid     = threadIdx.x;
  const int wave    = tid >> 5;
  const int lane    = tid & 31;
  const int rowBase = blockIdx.x * M_TILE;

  // ---- Stage X: stage 64x1024 f32 X tile as bf16 into LDS (read HBM once)
  {
    const float* Xt = X + (size_t)rowBase * D_DIM;
#pragma unroll 4
    for (int i = 0; i < (M_TILE * D_DIM / 4) / 256; ++i) {  // 64 iters
      const int f   = i * 256 + tid;  // float4 index, coalesced
      const int row = f >> 8;         // 256 float4 per row
      const int c4  = (f & 255) * 4;
      v4f x = *(const v4f*)(Xt + (size_t)row * D_DIM + c4);
      v4bf h;
      h[0] = (bf16_t)x[0]; h[1] = (bf16_t)x[1];
      h[2] = (bf16_t)x[2]; h[3] = (bf16_t)x[3];
      *(v4bf*)(Xs + row * QS_LD + c4) = h;
    }
  }
  __syncthreads();

  // ---- Stage A: Qs = bf16( Xs @ Wq ).  16 n-macros of width 64, 2/wave.
  for (int i = 0; i < 2; ++i) {
    const int n0 = (wave + NWAVES * i) * 64;
    v8f acc[16] = {};
    gemm_macrotile<D_DIM>(Xs, QS_LD, WqT, D_DIM, n0, lane, acc);
    const int col  = lane & 15;
    const int roff = (lane & 16) ? 8 : 0;
#pragma unroll
    for (int ni = 0; ni < 4; ++ni)
#pragma unroll
      for (int mi = 0; mi < 4; ++mi)
#pragma unroll
        for (int v = 0; v < 8; ++v)
          Qs[(mi * 16 + roff + v) * QS_LD + n0 + ni * 16 + col] =
              (bf16_t)acc[mi + 4 * ni][v];
  }
  __syncthreads();

  // ---- Stage B: Ss = SCALE * (Qs @ binding^T).  8 n-macros, 1/wave.
  // (Ss overwrites Xs, which is dead now.)
  {
    const int n0 = wave * 64;
    v8f acc[16] = {};
    gemm_macrotile<D_DIM>(Qs, QS_LD, BindB, D_DIM, n0, lane, acc);
    const int col  = lane & 15;
    const int roff = (lane & 16) ? 8 : 0;
#pragma unroll
    for (int ni = 0; ni < 4; ++ni)
#pragma unroll
      for (int mi = 0; mi < 4; ++mi)
#pragma unroll
        for (int v = 0; v < 8; ++v)
          Ss[(mi * 16 + roff + v) * SS_LD + n0 + ni * 16 + col] =
              acc[mi + 4 * ni][v] * SM_SCALE;
  }
  __syncthreads();

  // ---- Softmax over j=512 per row; write bf16 attn into As (over dead Qs).
  {
    const int row = tid >> 2;   // 0..63
    const int sub = tid & 3;    // 0..3, adjacent lanes of one wave
    const float* srow = Ss + row * SS_LD + sub * 128;
    float mx = -3.4e38f;
#pragma unroll 4
    for (int i = 0; i < 128; ++i) mx = fmaxf(mx, srow[i]);
    mx = fmaxf(mx, __shfl_xor(mx, 1));
    mx = fmaxf(mx, __shfl_xor(mx, 2));
    float sum = 0.0f;
#pragma unroll 4
    for (int i = 0; i < 128; ++i) sum += __expf(srow[i] - mx);
    sum += __shfl_xor(sum, 1);
    sum += __shfl_xor(sum, 2);
    const float inv = 1.0f / sum;
    bf16_t* arow = As + row * AS_LD + sub * 128;
#pragma unroll 4
    for (int i = 0; i < 128; ++i)
      arow[i] = (bf16_t)(__expf(srow[i] - mx) * inv);
  }
  __syncthreads();

  // ---- Stage C: Out = attn @ symbols.  16 n-macros, 2/wave, K = 512.
  for (int i = 0; i < 2; ++i) {
    const int n0 = (wave + NWAVES * i) * 64;
    v8f acc[16] = {};
    gemm_macrotile<J_DIM>(As, AS_LD, SymT, J_DIM, n0, lane, acc);
    const int col  = lane & 15;
    const int roff = (lane & 16) ? 8 : 0;
#pragma unroll
    for (int ni = 0; ni < 4; ++ni)
#pragma unroll
      for (int mi = 0; mi < 4; ++mi)
#pragma unroll
        for (int v = 0; v < 8; ++v)
          Out[(size_t)(rowBase + mi * 16 + roff + v) * D_DIM +
              n0 + ni * 16 + col] = acc[mi + 4 * ni][v];
  }
}

// ---------------------------------------------------------------------------
extern "C" void kernel_launch(void* const* d_in, const int* in_sizes, int n_in,
                              void* d_out, int out_size, void* d_ws, size_t ws_size,
                              hipStream_t stream) {
  const float* X       = (const float*)d_in[0];  // [8,4096,1024]
  const float* Wq      = (const float*)d_in[1];  // [1024,1024]
  const float* binding = (const float*)d_in[2];  // [512,1024]
  const float* symbols = (const float*)d_in[3];  // [512,1024]
  float* Out = (float*)d_out;

  // Workspace (bf16): WqT (2MB) | BindB (1MB) | SymT (1MB)
  bf16_t* WqT   = (bf16_t*)d_ws;
  bf16_t* BindB = WqT + (size_t)D_DIM * D_DIM;
  bf16_t* SymT  = BindB + (size_t)J_DIM * D_DIM;

  const int convElems = D_DIM * D_DIM;
  symret_convert<<<(convElems + 255) / 256, 256, 0, stream>>>(
      Wq, binding, symbols, WqT, BindB, SymT);

  const int rows = in_sizes[0] / D_DIM;          // 32768
  symret_fused<<<rows / M_TILE, 256, LDS_BYTES, stream>>>(
      X, WqT, BindB, SymT, Out);
}
